// TransformerEncoder_11871289606310
// MI455X (gfx1250) — matrix operand order repaired and checked by reading, not compiled, not run on hardware
//
#include <hip/hip_runtime.h>
#include <cstdint>
#include <cstddef>

#define DEV static __device__ __forceinline__

namespace {

constexpr int Bn = 8, Sn = 1024, Dn = 512, Hn = 8, KDn = 512, DMn = 4096, FFn = 2048;

typedef __attribute__((ext_vector_type(16))) __bf16 bf16x16;
typedef __attribute__((ext_vector_type(8)))  float  f32x8;
typedef __attribute__((ext_vector_type(4)))  unsigned int u32x4;

union Frag { u32x4 u4[2]; bf16x16 v; };

DEV unsigned short f2b(float f) {
  unsigned int u = __float_as_uint(f);
  u = (u + 0x7FFFu + ((u >> 16) & 1u)) >> 16;   // round-to-nearest-even
  return (unsigned short)u;
}

DEV f32x8 vzero() { f32x8 z = {0.f,0.f,0.f,0.f,0.f,0.f,0.f,0.f}; return z; }

// A-fragment (16 rows x 32 K, row-major src): half 0 -> K {0..15},
// half 1 -> K {16..31}; two 16-byte loads per lane.
DEV Frag load_a(const unsigned short* base, int ld, int k0, int mr, int half) {
  Frag f;
  const unsigned short* p = base + (size_t)mr * ld + k0 + half * 16;
  f.u4[0] = *(const u32x4*)p;
  f.u4[1] = *(const u32x4*)(p + 8);
  return f;
}

// B-fragment from [N][K]-stored operand: lane = column, half selects K 0-15 / 16-31.
DEV Frag load_b(const unsigned short* bt, int ld, int k0, int nc, int half) {
  Frag f;
  const unsigned short* p = bt + (size_t)nc * ld + k0 + half * 16;
  f.u4[0] = *(const u32x4*)p;
  f.u4[1] = *(const u32x4*)(p + 8);
  return f;
}

DEV f32x8 mma(const Frag& a, const Frag& b, f32x8 c) {
  return __builtin_amdgcn_wmma_f32_16x16x32_bf16(false, a.v, false, b.v,
                                                 (short)0, c, false, false);
}

} // namespace

// ---------------- conversion kernels ----------------
__global__ void k_cvt(const float* __restrict__ in, unsigned short* __restrict__ out, int n) {
  int i = blockIdx.x * 256 + threadIdx.x;
  if (i < n) out[i] = f2b(in[i]);
}

__global__ void k_cvt_t(const float* __restrict__ in, unsigned short* __restrict__ out,
                        int R, int C) {
  int i = blockIdx.x * 256 + threadIdx.x;
  if (i < R * C) {
    int r = i / C, c = i % C;
    out[(size_t)c * R + r] = f2b(in[i]);   // [K][N] -> [N][K]
  }
}

// ---------------- GEMM: C[128x128/block] = A(bf16,[M][K]) * B(bf16,[N][K]) ----------
// B panel staged through LDS (double buffered); 8 waves, each owns a 16x128 strip.
// MODE 0: +bias, store bf16 row-major
// MODE 1: +bias, store bf16 V-transposed [b][h][d][s]
// MODE 2: +bias +resid(f32), store f32
// MODE 3: +bias, relu, store bf16
#define BROW 40   // padded LDS row stride (halfwords) -> conflict-free frag reads

template<int MODE>
__global__ __launch_bounds__(256)
void k_gemm(const unsigned short* __restrict__ A, const unsigned short* __restrict__ BT,
            const float* __restrict__ bias, const float* __restrict__ resid,
            float* __restrict__ outF, unsigned short* __restrict__ outB,
            int M, int N, int K)
{
  (void)M;
  __shared__ unsigned short Bs[2][128 * BROW];

  const int t    = threadIdx.x;
  const int lane = t & 31;
  const int wave = t >> 5;
  const int half = lane >> 4;
  const int l16  = lane & 15;
  const int m0 = blockIdx.y * 128 + wave * 16;
  const int n0 = blockIdx.x * 128;

  f32x8 acc[8];
  #pragma unroll
  for (int j = 0; j < 8; ++j) acc[j] = vzero();

  const unsigned short* Abase = A + (size_t)m0 * K;

  // cooperative stage of one 128x32 B chunk: thread t -> row t>>1, 16-elem col (t&1)*16
  const int srow = t >> 1;
  const int scol = (t & 1) * 16;
  const unsigned short* Bsrc = BT + (size_t)(n0 + srow) * K + scol;

  // prologue stage k0 = 0
  {
    *(u32x4*)&Bs[0][srow * BROW + scol]     = *(const u32x4*)(Bsrc);
    *(u32x4*)&Bs[0][srow * BROW + scol + 8] = *(const u32x4*)(Bsrc + 8);
  }
  __syncthreads();

  for (int k0 = 0; k0 < K; k0 += 32) {
    const int cur = (k0 >> 5) & 1;
    if (k0 + 32 < K) {           // stage next chunk into the other buffer
      const unsigned short* src = Bsrc + k0 + 32;
      *(u32x4*)&Bs[cur ^ 1][srow * BROW + scol]     = *(const u32x4*)(src);
      *(u32x4*)&Bs[cur ^ 1][srow * BROW + scol + 8] = *(const u32x4*)(src + 8);
    }

    Frag a = load_a(Abase, K, k0, l16, half);
    #pragma unroll
    for (int j = 0; j < 8; ++j) {
      Frag b;
      const unsigned short* p = &Bs[cur][(16 * j + l16) * BROW + half * 16];
      b.u4[0] = *(const u32x4*)p;
      b.u4[1] = *(const u32x4*)(p + 8);
      acc[j] = mma(a, b, acc[j]);
    }
    __syncthreads();             // reads of cur done + writes of next visible
  }

  #pragma unroll
  for (int j = 0; j < 8; ++j) {
    #pragma unroll
    for (int r = 0; r < 8; ++r) {
      const int gm = m0 + r + 8 * half;      // C layout: VGPR r -> rows r / r+8
      const int gn = n0 + 16 * j + l16;
      float c = acc[j][r] + bias[gn];
      if (MODE == 2) {
        c += resid[(size_t)gm * N + gn];
        outF[(size_t)gm * N + gn] = c;
      } else if (MODE == 3) {
        outB[(size_t)gm * N + gn] = f2b(fmaxf(c, 0.f));
      } else if (MODE == 1) {
        // [b][h*KD+d][s] : (gm>>10)=batch, gm&1023=token, gn = h*KD+d
        outB[((size_t)(gm >> 10) * DMn + gn) * Sn + (gm & (Sn - 1))] = f2b(c);
      } else {
        outB[(size_t)gm * N + gn] = f2b(c);
      }
    }
  }
}

// ---------------- flash-style attention, one (b,h,16-query block) per workgroup ----
#define ATTN_SMEM (16 * 1024 * 4 + 16 * 1024 * 2 + 128 * 4 + 16 * 4)

__global__ __launch_bounds__(128)
void k_attn(const unsigned short* __restrict__ Qb, const unsigned short* __restrict__ Kb,
            const unsigned short* __restrict__ Vt, unsigned short* __restrict__ Ctx)
{
  extern __shared__ __align__(16) char smem[];
  float*          Ssc = (float*)smem;                      // 16 x 1024 f32 scores
  unsigned short* Pt  = (unsigned short*)(smem + 65536);   // 16 x 1024 bf16 probs
  float*          red = (float*)(smem + 65536 + 32768);    // 128 partials
  float*          rst = red + 128;                         // 16 row stats

  const int t    = threadIdx.x;
  const int lane = t & 31;
  const int wave = t >> 5;
  const int half = lane >> 4;
  const int l16  = lane & 15;

  const int qt = blockIdx.x & 63;
  const int h  = (blockIdx.x >> 6) & 7;
  const int b  = blockIdx.x >> 9;

  // keep the full 16x512 Q block in registers (16 fragments)
  const unsigned short* Qbase = Qb + (size_t)(b * Sn + qt * 16) * DMn + h * KDn;
  Frag qf[16];
  #pragma unroll
  for (int i = 0; i < 16; ++i) qf[i] = load_a(Qbase, DMn, i * 32, l16, half);

  // ---- pass 1: S = Q K^T / sqrt(KD); waves split the 64 key tiles ----
  const float scale = 0.044194173824159216f; // 1/sqrt(512)
  for (int kt = wave; kt < 64; kt += 4) {
    const unsigned short* Kbase = Kb + (size_t)(b * Sn + kt * 16) * DMn + h * KDn;
    f32x8 c = vzero();
    #pragma unroll
    for (int i = 0; i < 16; ++i) {
      Frag kb = load_b(Kbase, DMn, i * 32, l16, half);
      c = mma(qf[i], kb, c);
    }
    #pragma unroll
    for (int r = 0; r < 8; ++r)
      Ssc[(r + 8 * half) * 1024 + kt * 16 + l16] = c[r] * scale;
  }
  __syncthreads();

  // ---- softmax: 8 threads per row ----
  {
    const int rr = t >> 3;
    const int j0 = t & 7;
    float m = -3.4e38f;
    for (int j = j0; j < 1024; j += 8) m = fmaxf(m, Ssc[rr * 1024 + j]);
    red[t] = m;
    __syncthreads();
    if (t < 16) {
      float mm = red[t * 8];
      for (int i = 1; i < 8; ++i) mm = fmaxf(mm, red[t * 8 + i]);
      rst[t] = mm;
    }
    __syncthreads();
    const float mr = rst[rr];
    float s = 0.f;
    for (int j = j0; j < 1024; j += 8) {
      float e = __expf(Ssc[rr * 1024 + j] - mr);
      Ssc[rr * 1024 + j] = e;
      s += e;
    }
    red[t] = s;
    __syncthreads();
    if (t < 16) {
      float ss = 0.f;
      for (int i = 0; i < 8; ++i) ss += red[t * 8 + i];
      rst[t] = 1.0f / ss;
    }
    __syncthreads();
    const float inv = rst[rr];
    for (int j = j0; j < 1024; j += 8)
      Pt[rr * 1024 + j] = f2b(Ssc[rr * 1024 + j] * inv);
  }
  __syncthreads();

  // ---- pass 2: ctx = P @ V; each wave produces 8 of the 32 16-col tiles ----
  for (int nt = wave * 8; nt < wave * 8 + 8; ++nt) {
    const unsigned short* Vbase = Vt + ((size_t)(b * Hn + h) * KDn + nt * 16) * Sn;
    f32x8 c = vzero();
    for (int kt = 0; kt < 32; ++kt) {
      Frag pa = load_a(Pt, 1024, kt * 32, l16, half);       // from LDS
      Frag vb = load_b(Vbase, Sn, kt * 32, l16, half);      // key-contiguous
      c = mma(pa, vb, c);
    }
    unsigned short* Cb = Ctx + (size_t)(b * Sn + qt * 16) * DMn + h * KDn + nt * 16;
    #pragma unroll
    for (int r = 0; r < 8; ++r)
      Cb[(size_t)(r + 8 * half) * DMn + l16] = f2b(c[r]);
  }
}

// ---------------- LayerNorm: one wave per row ----------------
__global__ __launch_bounds__(256)
void k_ln(const float* __restrict__ x, const float* __restrict__ g,
          const float* __restrict__ be, float* __restrict__ outF,
          unsigned short* __restrict__ outB, int ncol)
{
  const int lane = threadIdx.x & 31;
  const int wave = threadIdx.x >> 5;
  const int row  = blockIdx.x * 8 + wave;
  const float* xr = x + (size_t)row * ncol;

  float s = 0.f;
  for (int j = lane; j < ncol; j += 32) s += xr[j];
  #pragma unroll
  for (int o = 16; o; o >>= 1) s += __shfl_xor(s, o, 32);
  const float mu = s / (float)ncol;

  float v = 0.f;
  for (int j = lane; j < ncol; j += 32) { float d = xr[j] - mu; v += d * d; }
  #pragma unroll
  for (int o = 16; o; o >>= 1) v += __shfl_xor(v, o, 32);
  const float rs = rsqrtf(v / (float)ncol + 1e-3f);

  for (int j = lane; j < ncol; j += 32) {
    float y = (xr[j] - mu) * rs * g[j] + be[j];
    outF[(size_t)row * ncol + j] = y;
    if (outB) outB[(size_t)row * ncol + j] = f2b(y);
  }
}

extern "C" void kernel_launch(void* const* d_in, const int* in_sizes, int n_in,
                              void* d_out, int out_size, void* d_ws, size_t ws_size,
                              hipStream_t stream)
{
  (void)in_sizes; (void)n_in; (void)out_size; (void)ws_size;
  const float* x   = (const float*)d_in[0];
  const float* w_q = (const float*)d_in[1];
  const float* b_q = (const float*)d_in[2];
  const float* w_k = (const float*)d_in[3];
  const float* b_k = (const float*)d_in[4];
  const float* w_v = (const float*)d_in[5];
  const float* b_v = (const float*)d_in[6];
  const float* w_o = (const float*)d_in[7];
  const float* b_o = (const float*)d_in[8];
  const float* w1  = (const float*)d_in[9];
  const float* b1  = (const float*)d_in[10];
  const float* w2  = (const float*)d_in[11];
  const float* b2  = (const float*)d_in[12];
  const float* g1  = (const float*)d_in[13];
  const float* be1 = (const float*)d_in[14];
  const float* g2  = (const float*)d_in[15];
  const float* be2 = (const float*)d_in[16];
  float* out = (float*)d_out;

  char* ws = (char*)d_ws;
  const size_t MB = 1024 * 1024;
  unsigned short* Xbf = (unsigned short*)(ws + 0);        //  8 MB
  unsigned short* WqT = (unsigned short*)(ws + 8 * MB);   //  4 MB
  unsigned short* WkT = (unsigned short*)(ws + 12 * MB);  //  4 MB
  unsigned short* WvT = (unsigned short*)(ws + 16 * MB);  //  4 MB
  unsigned short* WoT = (unsigned short*)(ws + 20 * MB);  //  4 MB
  unsigned short* W1T = (unsigned short*)(ws + 24 * MB);  //  2 MB
  unsigned short* W2T = (unsigned short*)(ws + 26 * MB);  //  2 MB
  unsigned short* Qb  = (unsigned short*)(ws + 28 * MB);  // 64 MB
  unsigned short* Kb  = (unsigned short*)(ws + 92 * MB);  // 64 MB
  unsigned short* Vt  = (unsigned short*)(ws + 156 * MB); // 64 MB
  unsigned short* Ctx = (unsigned short*)(ws + 220 * MB); // 64 MB (peak 284 MB)
  // Q/K regions are dead after attention -> reuse for epilogue buffers
  float*          T1  = (float*)(ws + 28 * MB);           // 16 MB
  float*          Y1f = (float*)(ws + 44 * MB);           // 16 MB
  unsigned short* Y1b = (unsigned short*)(ws + 60 * MB);  //  8 MB
  unsigned short* H1b = (unsigned short*)(ws + 68 * MB);  // 32 MB
  float*          T2  = (float*)(ws + 100 * MB);          // 16 MB

  const int MT = Bn * Sn; // 8192 tokens

  // stage activations + weights in bf16 (weights transposed to [N][K])
  k_cvt<<<(MT * Dn + 255) / 256, 256, 0, stream>>>(x, Xbf, MT * Dn);
  k_cvt_t<<<(Dn * DMn + 255) / 256, 256, 0, stream>>>(w_q, WqT, Dn, DMn);
  k_cvt_t<<<(Dn * DMn + 255) / 256, 256, 0, stream>>>(w_k, WkT, Dn, DMn);
  k_cvt_t<<<(Dn * DMn + 255) / 256, 256, 0, stream>>>(w_v, WvT, Dn, DMn);
  k_cvt_t<<<(DMn * Dn + 255) / 256, 256, 0, stream>>>(w_o, WoT, DMn, Dn);
  k_cvt_t<<<(Dn * FFn + 255) / 256, 256, 0, stream>>>(w1, W1T, Dn, FFn);
  k_cvt_t<<<(FFn * Dn + 255) / 256, 256, 0, stream>>>(w2, W2T, FFn, Dn);

  dim3 blk(256);
  // QKV projections (V stored transposed per head for PV fragment loads)
  k_gemm<0><<<dim3(DMn / 128, MT / 128), blk, 0, stream>>>(Xbf, WqT, b_q, nullptr, nullptr, Qb, MT, DMn, Dn);
  k_gemm<0><<<dim3(DMn / 128, MT / 128), blk, 0, stream>>>(Xbf, WkT, b_k, nullptr, nullptr, Kb, MT, DMn, Dn);
  k_gemm<1><<<dim3(DMn / 128, MT / 128), blk, 0, stream>>>(Xbf, WvT, b_v, nullptr, nullptr, Vt, MT, DMn, Dn);

  // attention
  k_attn<<<Bn * Hn * (Sn / 16), 128, ATTN_SMEM, stream>>>(Qb, Kb, Vt, Ctx);

  // output projection + residual, LN1
  k_gemm<2><<<dim3(Dn / 128, MT / 128), blk, 0, stream>>>(Ctx, WoT, b_o, x, T1, nullptr, MT, Dn, DMn);
  k_ln<<<MT / 8, 256, 0, stream>>>(T1, g1, be1, Y1f, Y1b, Dn);

  // FFN + residual, LN2 -> output
  k_gemm<3><<<dim3(FFn / 128, MT / 128), blk, 0, stream>>>(Y1b, W1T, b1, nullptr, nullptr, H1b, MT, FFn, Dn);
  k_gemm<2><<<dim3(Dn / 128, MT / 128), blk, 0, stream>>>(H1b, W2T, b2, Y1f, T2, nullptr, MT, Dn, FFn);
  k_ln<<<MT / 8, 256, 0, stream>>>(T2, g2, be2, out, nullptr, Dn);
}
